// MyModel_1_84189948936659
// MI455X (gfx1250) — matrix operand order repaired and checked by reading, not compiled, not run on hardware
//
#include <hip/hip_runtime.h>
#include <hip/hip_bf16.h>

// ---------------------------------------------------------------------------
// 3-layer tanh-RNN with LayerNorm, CDNA5 (gfx1250) WMMA implementation.
//   pack_w:     fp32 weights -> bf16 WMMA B-fragment layout in ws
//   embed_gemm: v_all[t][b][h] = inputs[b][t][:] @ embedding  (bf16)
//   rnn_scan:   4 persistent workgroups (16 batch rows each) run the full
//               T=1024 scan with state in LDS, weights streamed from L2 as
//               pre-packed bf16 fragments (software-pipelined K loop), v_t
//               prefetched via async global->LDS (ASYNCcnt) double buffer,
//               then classifier + softmax.
// ---------------------------------------------------------------------------

typedef __attribute__((ext_vector_type(16))) __bf16 v16bf;
typedef __attribute__((ext_vector_type(8)))  float  v8f;
typedef __attribute__((ext_vector_type(4)))  int    v4i;

#define WMMA_BF16(a, b, c) \
  __builtin_amdgcn_wmma_f32_16x16x32_bf16(false, (a), false, (b), (short)0, (c), false, false)

#if defined(__has_builtin)
#  if __has_builtin(__builtin_amdgcn_global_load_async_to_lds_b128)
#    define HAVE_ASYNC_B128 1
#  endif
#  if __has_builtin(__builtin_amdgcn_s_wait_asynccnt)
#    define HAVE_WAIT_ASYNC 1
#  endif
#endif

static constexpr int   B_    = 64;
static constexpr int   T_    = 1024;
static constexpr int   DH    = 512;
static constexpr int   DT    = 64;
static constexpr int   PITCH = 520;      // bf16 elems per LDS row (16B aligned, bank-friendly)
static constexpr size_t MATP = 262144;   // packed bf16 elems per 512x512 matrix

// A-fragment gather from LDS, 16-bit A 16x32 layout:
//   lane<16 : row M=lane,    K = ks*32 + {0..7, 16..23}
//   lane>=16: row M=lane-16, K = ks*32 + {8..15, 24..31}
__device__ __forceinline__ v16bf load_a_lds(const __bf16* p) {
  union { uint4 u[2]; v16bf v; } x;
  x.u[0] = *(const uint4*)(p);
  x.u[1] = *(const uint4*)(p + 16);
  return x.v;
}

// B-fragment from pre-packed global memory: 32 contiguous bytes per lane.
__device__ __forceinline__ v16bf load_b_glb(const __bf16* p) {
  union { uint4 u[2]; v16bf v; } x;
  x.u[0] = *(const uint4*)(p);
  x.u[1] = *(const uint4*)(p + 8);
  return x.v;
}

// One 16-byte async global->LDS copy (falls back to a sync copy when the
// async builtin is unavailable).  Builtin params are addrspace-typed int4*.
__device__ __forceinline__ void async_cp16(const __bf16* g, __bf16* l) {
#if defined(HAVE_ASYNC_B128)
  __builtin_amdgcn_global_load_async_to_lds_b128(
      (__attribute__((address_space(1))) v4i*)(g),
      (__attribute__((address_space(3))) v4i*)(l), 0, 0);
#else
  *(uint4*)l = *(const uint4*)g;
#endif
}

__device__ __forceinline__ void wait_async_done() {
#if defined(HAVE_ASYNC_B128)
#  if defined(HAVE_WAIT_ASYNC)
  __builtin_amdgcn_s_wait_asynccnt(0);
#  else
  asm volatile("s_wait_asynccnt 0x0" ::: "memory");
#  endif
#endif
}

// Pack W[K=512][N=ntiles*16] (row-major fp32) into bf16 WMMA B-fragments.
// B 32x16 16-bit layout: lane<16 holds K={0..7,16..23} of its column, lane>=16 K={8..15,24..31}.
__global__ __launch_bounds__(256) void pack_w(const float* __restrict__ W,
                                              __bf16* __restrict__ out, int ntiles) {
  int id = blockIdx.x * 256 + threadIdx.x;
  int total = ntiles * 16 * 32 * 16;
  if (id >= total) return;
  int e    = id & 15;
  int lane = (id >> 4) & 31;
  int f    = id >> 9;           // nt*16 + ks
  int ks   = f & 15;
  int nt   = f >> 4;
  int K = ks * 32 + ((lane >> 4) << 3) + (e & 7) + ((e >> 3) << 4);
  int N = nt * 16 + (lane & 15);
  out[id] = (__bf16)W[(size_t)K * (ntiles * 16) + N];
}

// v_all[t][b][h] = sum_e inputs[b][t][e] * embedding[e][h], stored bf16.
__global__ __launch_bounds__(256) void embed_gemm(const float* __restrict__ inputs,
                                                  const __bf16* __restrict__ embP,
                                                  __bf16* __restrict__ vall) {
  __shared__ __bf16 act[16][PITCH];
  const int rt = blockIdx.x;            // 0..4095
  const int t  = rt >> 2;
  const int b0 = (rt & 3) << 4;
  const int tid = threadIdx.x;

  for (int i = tid; i < 16 * DH; i += 256) {
    int r = i >> 9, e = i & 511;
    act[r][e] = (__bf16)inputs[((size_t)(b0 + r) * T_ + t) * DH + e];
  }
  __syncthreads();

  const int wave = tid >> 5, lane = tid & 31;
  const int hi = lane >> 4, m = lane & 15;
  const int rbase = hi * 8;

  for (int i = 0; i < 4; ++i) {
    const int nt = wave * 4 + i;
    v8f acc = {};
#pragma unroll
    for (int ks = 0; ks < 16; ++ks) {
      v16bf a = load_a_lds(&act[m][ks * 32 + 8 * hi]);
      v16bf b = load_b_glb(embP + ((size_t)(nt * 16 + ks) * 32 + lane) * 16);
      acc = WMMA_BF16(a, b, acc);
    }
    const int col = nt * 16 + m;
    for (int v = 0; v < 8; ++v)
      vall[((size_t)t * B_ + b0 + rbase + v) * DH + col] = (__bf16)acc[v];
  }
}

// Persistent scan kernel: grid=4 (16 batch rows each), 512 threads = 16 waves,
// each wave owns 2 N-tiles (32 of 512 hidden columns).
__global__ __launch_bounds__(512) void rnn_scan(
    const __bf16* __restrict__ vall,
    const __bf16* __restrict__ pw,      // packed [wx0, wh0, wx1, wh1, wx2, wh2]
    const float* __restrict__ bias0, const float* __restrict__ bias1,
    const float* __restrict__ bias2,
    const float* __restrict__ gamma, const float* __restrict__ beta,
    const __bf16* __restrict__ clfP, const float* __restrict__ clfb,
    float* __restrict__ out) {
  __shared__ __bf16 vbuf[2][16][PITCH];    // async double-buffered v_t
  __shared__ __bf16 act[16][PITCH];        // LayerNorm outputs (layer inputs 1,2)
  __shared__ __bf16 hst[3][16][PITCH];     // recurrent states (raw h, bf16)
  __shared__ float  stats[16][2];          // per-row sum / sumsq
  __shared__ float  lgts[16][DT];          // classifier logits

  const int tid  = threadIdx.x;
  const int wave = tid >> 5, lane = tid & 31;
  const int hi = lane >> 4, m = lane & 15;
  const int rbase = hi * 8;
  const int b0r = blockIdx.x * 16;

  const int nt0 = wave * 2;
  const int c0 = nt0 * 16 + m, c1 = c0 + 16;

  const float g0 = gamma[c0], g1 = gamma[c1];
  const float be0 = beta[c0], be1 = beta[c1];
  const float bl[3][2] = { {bias0[c0], bias0[c1]},
                           {bias1[c0], bias1[c1]},
                           {bias2[c0], bias2[c1]} };

  // Per-thread staging slot for v_t: one b128 (8 bf16) per thread covers only half of 16x512 bf16.
  const int sr = (tid * 16) >> 9;          // row 0..15
  const int se = (tid * 16) & 511;         // col (multiple of 16)

  {
    __bf16* hz = &hst[0][0][0];
    for (int i = tid; i < 3 * 16 * PITCH; i += 512) hz[i] = (__bf16)0.f;
  }
  // Kick off async staging of v_0.
  async_cp16(&vall[((size_t)0 * B_ + b0r + sr) * DH + se], &vbuf[0][sr][se]);
  __syncthreads();

  for (int t = 0; t < T_; ++t) {
    wait_async_done();                     // v_t resident in vbuf[t&1]
    __syncthreads();
    if (t + 1 < T_)                        // prefetch v_{t+1} into other buffer
      async_cp16(&vall[((size_t)(t + 1) * B_ + b0r + sr) * DH + se],
                 &vbuf[(t + 1) & 1][sr][se]);

    const __bf16 (*vsrc)[PITCH] = vbuf[t & 1];

    for (int l = 0; l < 3; ++l) {
      const __bf16 (*a1src)[PITCH] = (l == 0) ? vsrc : (const __bf16 (*)[PITCH])act;
      const __bf16* wx = pw + (size_t)(2 * l) * MATP;
      const __bf16* wh = pw + (size_t)(2 * l + 1) * MATP;
      // frag elem offset for (nt, ks): nt*8192 + ks*512 + lane*16
      const size_t fo = (size_t)nt0 * 8192 + (size_t)lane * 16;

      v8f acc0 = {}, acc1 = {};
      // Software-pipelined K loop: B-frags for ks+1 issued before WMMAs of ks.
      v16bf bx0 = load_b_glb(wx + fo);
      v16bf bh0 = load_b_glb(wh + fo);
      v16bf bx1 = load_b_glb(wx + fo + 8192);
      v16bf bh1 = load_b_glb(wh + fo + 8192);
#pragma unroll
      for (int ks = 0; ks < 16; ++ks) {
        v16bf nbx0 = bx0, nbh0 = bh0, nbx1 = bx1, nbh1 = bh1;
        if (ks < 15) {
          const size_t nfo = fo + (size_t)(ks + 1) * 512;
          nbx0 = load_b_glb(wx + nfo);
          nbh0 = load_b_glb(wh + nfo);
          nbx1 = load_b_glb(wx + nfo + 8192);
          nbh1 = load_b_glb(wh + nfo + 8192);
        }
        v16bf a1 = load_a_lds(&a1src[m][ks * 32 + 8 * hi]);
        v16bf a2 = load_a_lds(&hst[l][m][ks * 32 + 8 * hi]);
        acc0 = WMMA_BF16(a1, bx0, acc0);
        acc0 = WMMA_BF16(a2, bh0, acc0);
        acc1 = WMMA_BF16(a1, bx1, acc1);
        acc1 = WMMA_BF16(a2, bh1, acc1);
        bx0 = nbx0; bh0 = nbh0; bx1 = nbx1; bh1 = nbh1;
      }
      __syncthreads();                     // done reading act/vbuf & hst[l]
      if (tid < 32) (&stats[0][0])[tid] = 0.f;
      __syncthreads();

      // tanh, write new state, accumulate LN stats (in-wave tree + one atomic).
      float hv0[8], hv1[8], s0[8], s1[8];
      for (int v = 0; v < 8; ++v) {
        float x0 = tanhf(acc0[v] + bl[l][0]);
        float x1 = tanhf(acc1[v] + bl[l][1]);
        hv0[v] = x0; hv1[v] = x1;
        hst[l][rbase + v][c0] = (__bf16)x0;
        hst[l][rbase + v][c1] = (__bf16)x1;
        s0[v] = x0 + x1;
        s1[v] = x0 * x0 + x1 * x1;
      }
      for (int off = 1; off < 16; off <<= 1)
        for (int v = 0; v < 8; ++v) {
          s0[v] += __shfl_xor(s0[v], off, 32);
          s1[v] += __shfl_xor(s1[v], off, 32);
        }
      if (m == 0)
        for (int v = 0; v < 8; ++v) {
          atomicAdd(&stats[rbase + v][0], s0[v]);
          atomicAdd(&stats[rbase + v][1], s1[v]);
        }
      __syncthreads();

      // LayerNorm -> activation buffer for next layer.
      for (int v = 0; v < 8; ++v) {
        const int r = rbase + v;
        float mu  = stats[r][0] * (1.f / 512.f);
        float var = stats[r][1] * (1.f / 512.f) - mu * mu;
        float inv = rsqrtf(var + 1e-5f);
        act[r][c0] = (__bf16)((hv0[v] - mu) * inv * g0 + be0);
        act[r][c1] = (__bf16)((hv1[v] - mu) * inv * g1 + be1);
      }
      __syncthreads();
    }
  }

  // Classifier: logits = o2 @ clf_w + clf_b  ([16,512]@[512,64]).
  // 16 waves: 4 N-tiles x 4-way K-split, partials merged via LDS atomics.
  for (int i = tid; i < 16 * DT; i += 512) lgts[i >> 6][i & 63] = clfb[i & 63];
  __syncthreads();
  {
    const int nt = wave & 3;
    const int kq = wave >> 2;
    v8f acc = {};
    for (int ks = kq * 4; ks < kq * 4 + 4; ++ks) {
      v16bf a = load_a_lds(&act[m][ks * 32 + 8 * hi]);
      v16bf b = load_b_glb(clfP + ((size_t)(nt * 16 + ks) * 32 + lane) * 16);
      acc = WMMA_BF16(a, b, acc);
    }
    const int cc = nt * 16 + m;
    for (int v = 0; v < 8; ++v) atomicAdd(&lgts[rbase + v][cc], acc[v]);
  }
  __syncthreads();

  // Softmax per row (64 columns), one thread per row.
  if (tid < 16) {
    float mx = -3.4e38f;
    for (int c = 0; c < DT; ++c) mx = fmaxf(mx, lgts[tid][c]);
    float s = 0.f;
    for (int c = 0; c < DT; ++c) s += __expf(lgts[tid][c] - mx);
    float inv = 1.f / s;
    for (int c = 0; c < DT; ++c)
      out[(size_t)(b0r + tid) * DT + c] = __expf(lgts[tid][c] - mx) * inv;
  }
}

extern "C" void kernel_launch(void* const* d_in, const int* in_sizes, int n_in,
                              void* d_out, int out_size, void* d_ws, size_t ws_size,
                              hipStream_t stream) {
  (void)in_sizes; (void)n_in; (void)out_size; (void)ws_size;

  const float* inputs = (const float*)d_in[0];

  // Workspace: [ v_all bf16 (64 MB) | packed weights bf16 (~3.7 MB) ]
  __bf16* vall = (__bf16*)d_ws;
  __bf16* pack = (__bf16*)((char*)d_ws + (size_t)T_ * B_ * DH * sizeof(__bf16));

  // Pack order: embedding, wx0, wh0, wx1, wh1, wx2, wh2, clf_w.
  const float* srcs[8] = {
      (const float*)d_in[1],  (const float*)d_in[2], (const float*)d_in[3],
      (const float*)d_in[5],  (const float*)d_in[6], (const float*)d_in[8],
      (const float*)d_in[9],  (const float*)d_in[13] };
  const int ntl[8] = {32, 32, 32, 32, 32, 32, 32, 4};
  size_t off = 0;
  for (int i = 0; i < 8; ++i) {
    int total = ntl[i] * 8192;
    pack_w<<<(total + 255) / 256, 256, 0, stream>>>(srcs[i], pack + off, ntl[i]);
    off += (size_t)total;
  }

  embed_gemm<<<4096, 256, 0, stream>>>(inputs, pack /* embedding at offset 0 */, vall);

  rnn_scan<<<4, 512, 0, stream>>>(
      vall, pack + MATP /* wx0.. */,
      (const float*)d_in[4], (const float*)d_in[7], (const float*)d_in[10],
      (const float*)d_in[11], (const float*)d_in[12],
      pack + 7 * MATP /* clf_w */, (const float*)d_in[14],
      (float*)d_out);
}
